// RelationalGraphBuilder_82351702933796
// MI455X (gfx1250) — compile-verified, matrix operand order checked by reading, not run on hardware
//
#include <hip/hip_runtime.h>
#include <math.h>

// RelationalGraphBuilder for MI455X (gfx1250, wave32).
// Big GEMMs (65536x1536x512 and 65536x512x512) via v_wmma_f32_16x16x32_f16.
// M=32 tile per workgroup: every B fragment feeds 2 WMMAs (halves L2 weight
// traffic vs M=16), 8 independent accumulator chains hide memory latency.

constexpr int TT  = 256;    // tokens
constexpr int HH  = 512;    // hidden
constexpr int H3  = 1536;   // 3*H
constexpr int NE  = 8;      // edge types

typedef __attribute__((ext_vector_type(16))) _Float16 v16h;
typedef __attribute__((ext_vector_type(8)))  float    v8f;

struct U4 { unsigned int a, b, c, d; };      // POD 16-byte chunk
union Frag { v16h v; U4 q[2]; };

__device__ inline float rowred8(float v) {
#pragma unroll
  for (int m = 4; m >= 1; m >>= 1) v += __shfl_xor(v, m, 8);
  return v;
}

// ---------------------------------------------------------------------------
// Pre-swizzle a row-major weight w[O][K] (O=512) into per-lane WMMA B-fragment
// order (f16). Fragment linear index: ((ks*32 + ot)*32 + lane)*16 + j,
// k = ks*32 + 16*(j>>3) + 8*(lane>>4) + (j&7),  o = ot*16 + (lane&15).
// ---------------------------------------------------------------------------
__global__ __launch_bounds__(256) void prep_frags_kernel(
    const float* __restrict__ w, _Float16* __restrict__ frag, int K) {
  int idx  = blockIdx.x * 256 + threadIdx.x;
  int j    = idx & 15;
  int lane = (idx >> 4) & 31;
  int rest = idx >> 9;               // ks*32 + ot   (O/16 == 32)
  int ot   = rest & 31;
  int ks   = rest >> 5;
  int k    = ks * 32 + ((j >> 3) << 4) + ((lane >> 4) << 3) + (j & 7);
  int o    = (ot << 4) + (lane & 15);
  frag[idx] = (_Float16)w[(size_t)o * K + k];
}

// ---------------------------------------------------------------------------
// Fused edge pipeline: one block = 32 rows (n = t*256+s) x full H output.
// LDS overlay (96KB total):
//   phase 1-2:  s_ef  = f16[32*1536]              (96KB)
//   phase 2-5:  s_xf  = f32[32*512]  @ offset 0   (64KB)
//   phase 3-4:  s_w   = f16[32*512]  @ offset 64K (32KB)
// ---------------------------------------------------------------------------
__global__ __launch_bounds__(256) void edges_kernel(
    const float* __restrict__ nodes,
    const float* __restrict__ es_g1, const float* __restrict__ es_b1,
    const float* __restrict__ es_bb1,
    const float* __restrict__ es_g2, const float* __restrict__ es_b2,
    const float* __restrict__ es_w2, const float* __restrict__ es_bb2,
    const float* __restrict__ edge_emb,
    const float* __restrict__ ee_b, const float* __restrict__ ee_g,
    const float* __restrict__ ee_be,
    const _Float16* __restrict__ w1f,   // es_w1 fragments (K=1536)
    const _Float16* __restrict__ w2f,   // ee_w fragments  (K=512)
    float* __restrict__ edges_out)      // [T*T][H]
{
  __shared__ __align__(16) unsigned char smem[32 * H3 * 2];   // 98304 B
  _Float16* s_ef = (_Float16*)smem;                 // 32 x 1536 f16
  float*    s_xf = (float*)smem;                    // 32 x 512  f32
  _Float16* s_w  = (_Float16*)(smem + 65536);       // 32 x 512  f16

  const int tid = threadIdx.x;
  const int r   = tid >> 3;          // row in tile 0..31
  const int c   = tid & 7;           // column group 0..7
  const int n   = blockIdx.x * 32 + r;
  const int t   = n >> 8;
  const int s   = n & 255;
  const float* nt = nodes + t * HH;
  const float* ns = nodes + s * HH;

  // ---- phase 1: ef row (concat[q, k, q*k/8]) + LayerNorm over 3H -> f16 LDS
  float sum = 0.f, ssq = 0.f;
  for (int k = c * 192; k < c * 192 + 192; ++k) {
    float v;
    if (k < HH)          v = nt[k];
    else if (k < 2 * HH) v = ns[k - HH];
    else                 v = nt[k - 2 * HH] * ns[k - 2 * HH] * 0.125f;
    sum += v; ssq += v * v;
  }
  sum = rowred8(sum); ssq = rowred8(ssq);
  float mean = sum * (1.f / H3);
  float rstd = rsqrtf(ssq * (1.f / H3) - mean * mean + 1e-5f);
  for (int k = c * 192; k < c * 192 + 192; ++k) {
    float v;
    if (k < HH)          v = nt[k];
    else if (k < 2 * HH) v = ns[k - HH];
    else                 v = nt[k - 2 * HH] * ns[k - 2 * HH] * 0.125f;
    s_ef[r * H3 + k] = (_Float16)((v - mean) * rstd * es_g1[k] + es_b1[k]);
  }
  __syncthreads();

  // ---- phase 2: GEMM1 [32x1536]@[1536x512] via WMMA f16, 2 M-tiles/wave ----
  const int wave = tid >> 5;
  const int lane = tid & 31;
  const int m    = lane & 15;
  const int kb   = (lane >> 4) << 3;       // 0 or 8
  const int mrow = (lane >> 4) << 3;       // D-row offset 0 or 8
  v8f acc[8];                              // [mt*4 + i]
#pragma unroll
  for (int i = 0; i < 8; ++i) acc[i] = (v8f){0,0,0,0,0,0,0,0};
  for (int ks = 0; ks < 48; ++ks) {
    Frag a0, a1;
    const _Float16* ap0 = s_ef + m * H3 + ks * 32;
    const _Float16* ap1 = s_ef + (m + 16) * H3 + ks * 32;
    a0.q[0] = *(const U4*)(ap0 + kb);
    a0.q[1] = *(const U4*)(ap0 + 16 + kb);
    a1.q[0] = *(const U4*)(ap1 + kb);
    a1.q[1] = *(const U4*)(ap1 + 16 + kb);
    const _Float16* bp = w1f + ((size_t)(ks * 32 + wave * 4) * 32 + lane) * 16;
    Frag b[4];
#pragma unroll
    for (int i = 0; i < 4; ++i) {
      b[i].q[0] = *(const U4*)(bp + i * 512);
      b[i].q[1] = *(const U4*)(bp + i * 512 + 8);
    }
#pragma unroll
    for (int i = 0; i < 4; ++i) {
      acc[i]     = __builtin_amdgcn_wmma_f32_16x16x32_f16(false, a0.v, false, b[i].v, (short)0, acc[i],     false, false);
      acc[4 + i] = __builtin_amdgcn_wmma_f32_16x16x32_f16(false, a1.v, false, b[i].v, (short)0, acc[4 + i], false, false);
    }
  }
  __syncthreads();   // all GEMM1 reads of s_ef done; s_xf overlays s_ef
  // +bias, exact GELU, scatter to fp32 LDS
#pragma unroll
  for (int rr = 0; rr < 8; ++rr) {
    int mm = rr + mrow;
    int o0 = wave * 64 + (lane & 15);
#pragma unroll
    for (int i = 0; i < 4; ++i) {
      int o = o0 + i * 16;
      float v0 = acc[i][rr] + es_bb1[o];
      float v1 = acc[4 + i][rr] + es_bb1[o];
      s_xf[mm * HH + o]        = 0.5f * v0 * (1.f + erff(v0 * 0.70710678f));
      s_xf[(mm + 16) * HH + o] = 0.5f * v1 * (1.f + erff(v1 * 0.70710678f));
    }
  }
  __syncthreads();

  // ---- phase 3: LN(H) -> logits(E=8) -> softmax -> probs@edge_emb -> f16 LDS
  {
    float* xr = s_xf + r * HH;
    float s1 = 0.f, s2 = 0.f;
    for (int k = c * 64; k < c * 64 + 64; ++k) { float v = xr[k]; s1 += v; s2 += v * v; }
    s1 = rowred8(s1); s2 = rowred8(s2);
    float mu = s1 * (1.f / HH);
    float rs = rsqrtf(s2 * (1.f / HH) - mu * mu + 1e-5f);
    float lg[NE];
#pragma unroll
    for (int e = 0; e < NE; ++e) lg[e] = 0.f;
    for (int k = c * 64; k < c * 64 + 64; ++k) {
      float xn = (xr[k] - mu) * rs * es_g2[k] + es_b2[k];
#pragma unroll
      for (int e = 0; e < NE; ++e) lg[e] += xn * es_w2[e * HH + k];
    }
#pragma unroll
    for (int e = 0; e < NE; ++e) lg[e] = rowred8(lg[e]) + es_bb2[e];
    float mx = lg[0];
#pragma unroll
    for (int e = 1; e < NE; ++e) mx = fmaxf(mx, lg[e]);
    float se = 0.f;
#pragma unroll
    for (int e = 0; e < NE; ++e) { lg[e] = expf(lg[e] - mx); se += lg[e]; }
    float inv = 1.f / se;
#pragma unroll
    for (int e = 0; e < NE; ++e) lg[e] *= inv;
    for (int k = c * 64; k < c * 64 + 64; ++k) {
      float wv = 0.f;
#pragma unroll
      for (int e = 0; e < NE; ++e) wv += lg[e] * edge_emb[e * HH + k];
      s_w[r * HH + k] = (_Float16)wv;
    }
  }
  __syncthreads();

  // ---- phase 4: GEMM2 [32x512]@[512x512] via WMMA f16 ----
  v8f d[8];
#pragma unroll
  for (int i = 0; i < 8; ++i) d[i] = (v8f){0,0,0,0,0,0,0,0};
  for (int ks = 0; ks < 16; ++ks) {
    Frag a0, a1;
    const _Float16* ap0 = s_w + m * HH + ks * 32;
    const _Float16* ap1 = s_w + (m + 16) * HH + ks * 32;
    a0.q[0] = *(const U4*)(ap0 + kb);
    a0.q[1] = *(const U4*)(ap0 + 16 + kb);
    a1.q[0] = *(const U4*)(ap1 + kb);
    a1.q[1] = *(const U4*)(ap1 + 16 + kb);
    const _Float16* bp = w2f + ((size_t)(ks * 32 + wave * 4) * 32 + lane) * 16;
    Frag b[4];
#pragma unroll
    for (int i = 0; i < 4; ++i) {
      b[i].q[0] = *(const U4*)(bp + i * 512);
      b[i].q[1] = *(const U4*)(bp + i * 512 + 8);
    }
#pragma unroll
    for (int i = 0; i < 4; ++i) {
      d[i]     = __builtin_amdgcn_wmma_f32_16x16x32_f16(false, a0.v, false, b[i].v, (short)0, d[i],     false, false);
      d[4 + i] = __builtin_amdgcn_wmma_f32_16x16x32_f16(false, a1.v, false, b[i].v, (short)0, d[4 + i], false, false);
    }
  }
  // s_xf readers all finished before the pre-GEMM2 barrier; safe to overwrite.
#pragma unroll
  for (int rr = 0; rr < 8; ++rr) {
    int mm = rr + mrow;
    int o0 = wave * 64 + (lane & 15);
#pragma unroll
    for (int i = 0; i < 4; ++i) {
      int o = o0 + i * 16;
      s_xf[mm * HH + o]        = d[i][rr]     + ee_b[o];
      s_xf[(mm + 16) * HH + o] = d[4 + i][rr] + ee_b[o];
    }
  }
  __syncthreads();

  // ---- phase 5: final LN and write edges ----
  {
    float* yr = s_xf + r * HH;
    float s1 = 0.f, s2 = 0.f;
    for (int k = c * 64; k < c * 64 + 64; ++k) { float v = yr[k]; s1 += v; s2 += v * v; }
    s1 = rowred8(s1); s2 = rowred8(s2);
    float mu = s1 * (1.f / HH);
    float rs = rsqrtf(s2 * (1.f / HH) - mu * mu + 1e-5f);
    float* outp = edges_out + (size_t)n * HH;
    for (int k = c * 64; k < c * 64 + 64; ++k)
      outp[k] = (yr[k] - mu) * rs * ee_g[k] + ee_be[k];
  }
}

// ---------------------------------------------------------------------------
// Attention layers (tiny: <1.5% of total FLOPs) — fp32.
// ---------------------------------------------------------------------------
__global__ __launch_bounds__(256) void copy_kernel(
    const float* __restrict__ src, float* __restrict__ dst, int nelem) {
  int i = blockIdx.x * 256 + threadIdx.x;
  if (i < nelem) dst[i] = src[i];
}

__global__ __launch_bounds__(256) void qkv_kernel(
    const float* __restrict__ cur, const float* __restrict__ inw,
    const float* __restrict__ inb, float* __restrict__ qkv) {
  int t = blockIdx.x;
  const float* cr = cur + t * HH;
#pragma unroll
  for (int i = 0; i < 6; ++i) {
    int o = threadIdx.x + 256 * i;
    const float* wr = inw + (size_t)o * HH;
    float acc = inb[o];
    for (int k = 0; k < HH; ++k) acc = fmaf(cr[k], wr[k], acc);
    qkv[t * H3 + o] = acc;
  }
}

__global__ __launch_bounds__(256) void attn_kernel(
    const float* __restrict__ cur, const float* __restrict__ qkv,
    const float* __restrict__ edges, float* __restrict__ obuf) {
  __shared__ float sc[TT];
  __shared__ float red[TT];
  int t = blockIdx.x, h = blockIdx.y, si = threadIdx.x;
  const float* qp = qkv + t * H3 + h * 64;
  const float* kp = qkv + si * H3 + HH + h * 64;
  const float* cp = cur + t * HH + h * 64;
  const float* ep = edges + ((size_t)si * TT + t) * HH + h * 64;
  float dq = 0.f, db = 0.f;
  for (int k = 0; k < 64; ++k) dq = fmaf(qp[k], kp[k], dq);
  for (int k = 0; k < 64; ++k) db = fmaf(cp[k], ep[k], db);
  float score = (dq + db) * 0.125f;
  red[si] = score; __syncthreads();
  for (int st = 128; st > 0; st >>= 1) {
    if (si < st) red[si] = fmaxf(red[si], red[si + st]);
    __syncthreads();
  }
  float mx = red[0]; __syncthreads();
  float ex = expf(score - mx);
  red[si] = ex; __syncthreads();
  for (int st = 128; st > 0; st >>= 1) {
    if (si < st) red[si] += red[si + st];
    __syncthreads();
  }
  sc[si] = ex / red[0];
  __syncthreads();
  if (si < 64) {
    float acc = 0.f;
    for (int s2 = 0; s2 < TT; ++s2)
      acc = fmaf(sc[s2], qkv[s2 * H3 + 2 * HH + h * 64 + si], acc);
    obuf[t * HH + h * 64 + si] = acc;
  }
}

__global__ __launch_bounds__(256) void proj_ln_kernel(
    const float* __restrict__ obuf, const float* __restrict__ outw,
    const float* __restrict__ outb, const float* __restrict__ g,
    const float* __restrict__ b, float* __restrict__ cur) {
  __shared__ float y[HH];
  __shared__ float red[256];
  int t = blockIdx.x, tid = threadIdx.x;
  const float* orow = obuf + t * HH;
#pragma unroll
  for (int i = 0; i < 2; ++i) {
    int j = tid + 256 * i;
    const float* wr = outw + (size_t)j * HH;
    float acc = outb[j];
    for (int k = 0; k < HH; ++k) acc = fmaf(orow[k], wr[k], acc);
    y[j] = cur[t * HH + j] + acc;
  }
  __syncthreads();
  float p1 = y[tid] + y[tid + 256];
  float p2 = y[tid] * y[tid] + y[tid + 256] * y[tid + 256];
  red[tid] = p1; __syncthreads();
  for (int st = 128; st > 0; st >>= 1) { if (tid < st) red[tid] += red[tid + st]; __syncthreads(); }
  float mu = red[0] * (1.f / HH); __syncthreads();
  red[tid] = p2; __syncthreads();
  for (int st = 128; st > 0; st >>= 1) { if (tid < st) red[tid] += red[tid + st]; __syncthreads(); }
  float rs = rsqrtf(red[0] * (1.f / HH) - mu * mu + 1e-5f);
#pragma unroll
  for (int i = 0; i < 2; ++i) {
    int j = tid + 256 * i;
    cur[t * HH + j] = (y[j] - mu) * rs * g[j] + b[j];
  }
}

__global__ __launch_bounds__(256) void final_kernel(
    const float* __restrict__ nodes, const float* __restrict__ cur,
    const float* __restrict__ enh, float* __restrict__ out) {
  int i = blockIdx.x * 256 + threadIdx.x;
  float sg = 1.f / (1.f + expf(-enh[0]));
  out[i] = nodes[i] + sg * cur[i];
}

// ---------------------------------------------------------------------------
extern "C" void kernel_launch(void* const* d_in, const int* in_sizes, int n_in,
                              void* d_out, int out_size, void* d_ws, size_t ws_size,
                              hipStream_t stream) {
  (void)in_sizes; (void)n_in; (void)out_size; (void)ws_size;
  const float* nodes    = (const float*)d_in[0];
  const float* edge_emb = (const float*)d_in[1];
  const float* ee_w     = (const float*)d_in[2];
  const float* ee_b     = (const float*)d_in[3];
  const float* ee_g     = (const float*)d_in[4];
  const float* ee_be    = (const float*)d_in[5];
  const float* es_g1    = (const float*)d_in[6];
  const float* es_b1    = (const float*)d_in[7];
  const float* es_w1    = (const float*)d_in[8];
  const float* es_bb1   = (const float*)d_in[9];
  const float* es_g2    = (const float*)d_in[10];
  const float* es_b2    = (const float*)d_in[11];
  const float* es_w2    = (const float*)d_in[12];
  const float* es_bb2   = (const float*)d_in[13];
  const float* enh      = (const float*)d_in[26];

  char* ws = (char*)d_ws;
  _Float16* w1f  = (_Float16*)(ws);            // 786432 halves = 1,572,864 B
  _Float16* w2f  = (_Float16*)(ws + 1572864);  // 262144 halves =   524,288 B
  float*    cur  = (float*)(ws + 2097152);     // 131072 f32
  float*    qkv  = (float*)(ws + 2621440);     // 393216 f32
  float*    obuf = (float*)(ws + 4194304);     // 131072 f32

  float* edges_out = (float*)d_out;            // 33,554,432 floats
  float* final_out = edges_out + 33554432;

  prep_frags_kernel<<<3072, 256, 0, stream>>>(es_w1, w1f, H3);
  prep_frags_kernel<<<1024, 256, 0, stream>>>(ee_w, w2f, HH);
  edges_kernel<<<2048, 256, 0, stream>>>(nodes, es_g1, es_b1, es_bb1,
                                         es_g2, es_b2, es_w2, es_bb2,
                                         edge_emb, ee_b, ee_g, ee_be,
                                         w1f, w2f, edges_out);
  copy_kernel<<<512, 256, 0, stream>>>(nodes, cur, TT * HH);
  for (int l = 0; l < 2; ++l) {
    const float* inw  = (const float*)d_in[14 + 6 * l];
    const float* inb  = (const float*)d_in[15 + 6 * l];
    const float* outw = (const float*)d_in[16 + 6 * l];
    const float* outb = (const float*)d_in[17 + 6 * l];
    const float* g    = (const float*)d_in[18 + 6 * l];
    const float* b    = (const float*)d_in[19 + 6 * l];
    qkv_kernel<<<256, 256, 0, stream>>>(cur, inw, inb, qkv);
    attn_kernel<<<dim3(256, 8), 256, 0, stream>>>(cur, qkv, edges_out, obuf);
    proj_ln_kernel<<<256, 256, 0, stream>>>(obuf, outw, outb, g, b, cur);
  }
  final_kernel<<<512, 256, 0, stream>>>(nodes, cur, enh, final_out);
}
